// GCN_20272245637548
// MI455X (gfx1250) — compile-verified, matrix operand order checked by reading, not run on hardware
//
#include <hip/hip_runtime.h>

// ---------- types ----------
typedef __attribute__((ext_vector_type(16))) __bf16    bf16x16;
typedef __attribute__((ext_vector_type(8)))  float     f32x8;
typedef __attribute__((ext_vector_type(4)))  unsigned  uint4_;

union FragB { bf16x16 v; unsigned short u[16]; uint4_ q[2]; };

__device__ __forceinline__ unsigned short f2bf_bits(float f) {
  __bf16 b = (__bf16)f;                     // native cvt, RNE
  unsigned short s;
  __builtin_memcpy(&s, &b, 2);
  return s;
}

// order-preserving float <-> uint map for atomicMax-based segment_max
__device__ __forceinline__ unsigned enc_f(float f) {
  unsigned u = __float_as_uint(f);
  return (u & 0x80000000u) ? ~u : (u | 0x80000000u);
}
__device__ __forceinline__ float dec_f(unsigned u) {
  unsigned b = (u & 0x80000000u) ? (u & 0x7FFFFFFFu) : ~u;
  return __uint_as_float(b);
}
#define ENC_NEG_INF 0x007FFFFFu   // enc(-inf)

// ---------- small utility kernels ----------
__global__ void init_enc_kernel(unsigned* p, int n) {
  int i = blockIdx.x * blockDim.x + threadIdx.x;
  if (i < n) p[i] = ENC_NEG_INF;
}
__global__ void decode_kernel(unsigned* p, int n) {
  int i = blockIdx.x * blockDim.x + threadIdx.x;
  if (i < n) { float f = dec_f(p[i]); p[i] = __float_as_uint(f); }
}
__global__ void deg_init_kernel(int* deg, int n) {
  int i = blockIdx.x * blockDim.x + threadIdx.x;
  if (i < n) deg[i] = 1;                     // self loop
}
__global__ void deg_count_kernel(const long long* __restrict__ dst, int* deg, int E) {
  int e = blockIdx.x * blockDim.x + threadIdx.x;
  if (e < E) atomicAdd(&deg[(int)dst[e]], 1);
}
__global__ void dinv_kernel(const int* __restrict__ deg, float* __restrict__ dinv, int n) {
  int i = blockIdx.x * blockDim.x + threadIdx.x;
  if (i < n) dinv[i] = rsqrtf((float)deg[i]);
}
// W [K][N] f32 (row-major) -> Wt [N][K] bf16
__global__ void transpose_bf16_kernel(const float* __restrict__ W,
                                      unsigned short* __restrict__ Wt, int K, int N) {
  int tid = blockIdx.x * blockDim.x + threadIdx.x;
  if (tid >= K * N) return;
  int k = tid / N, n = tid - k * N;
  Wt[(size_t)n * K + k] = f2bf_bits(W[tid]);
}
// f32 -> bf16 elementwise
__global__ void cvt_bf16_kernel(const float* __restrict__ in,
                                unsigned short* __restrict__ out, int n) {
  int i = blockIdx.x * blockDim.x + threadIdx.x;
  if (i < n) out[i] = f2bf_bits(in[i]);
}

// ---------- WMMA GEMM: C[M,N] = A[M,K](bf16) x Bt[N,K](bf16)^T ----------
// block = 256 threads = 8 waves; wave w -> rows [blk*128 + w*16, +16),
// cols [blockIdx.y*(NT*16), +NT*16). NT is a compile-time constant so the
// K-loop is straight-line, accumulators pinned.
// All loads of a K-step are issued before the first WMMA so the fragments
// occupy distinct registers and the s_wait_loadcnt waits are staged
// (loads overlap earlier WMMAs instead of serializing).
// Out-of-range rows: A pointer clamped to row 0 (C row i depends only on
// A row i; garbage rows are dropped by the store guard) — keeps the hot
// loop free of EXEC manipulation.
template <int NT>
__launch_bounds__(256)
__global__ void gemm_bf16_kernel(const unsigned short* __restrict__ A,
                                 const unsigned short* __restrict__ Bt,
                                 const float* __restrict__ rowscale,   // may be null
                                 const float* __restrict__ bias,       // may be null
                                 float* __restrict__ C,                // f32 out (or null)
                                 unsigned short* __restrict__ C16,     // bf16 out (or null)
                                 int M, int K, int N) {
  const int lane = threadIdx.x & 31;
  const int wave = threadIdx.x >> 5;
  const int rowTile = blockIdx.x * 128 + wave * 16;
  const int colBase = blockIdx.y * (NT * 16);
  const int m  = lane & 15;
  const int hi = lane >> 4;                  // half-wave select (ISA K grouping)
  const int r  = rowTile + m;
  const int rc = (r < M) ? r : 0;            // clamped row for loads
  const unsigned short* __restrict__ Arow = A + (size_t)rc * K + hi * 8;

  f32x8 acc[NT];
  #pragma unroll
  for (int t = 0; t < NT; ++t)
    #pragma unroll
    for (int i = 0; i < 8; ++i) acc[t][i] = 0.0f;

  for (int k0 = 0; k0 < K; k0 += 32) {
    // ---- issue ALL loads first (distinct fragment registers) ----
    FragB af, bfr[NT];
    af.q[0] = *(const uint4_*)(Arow + k0);
    af.q[1] = *(const uint4_*)(Arow + k0 + 16);
    #pragma unroll
    for (int t = 0; t < NT; ++t) {
      const unsigned short* p = Bt + (size_t)(colBase + t * 16 + m) * K + k0 + hi * 8;
      bfr[t].q[0] = *(const uint4_*)(p);
      bfr[t].q[1] = *(const uint4_*)(p + 16);
    }
    // ---- then the matrix ops ----
    #pragma unroll
    for (int t = 0; t < NT; ++t) {
      acc[t] = __builtin_amdgcn_wmma_f32_16x16x32_bf16(
          false, af.v, false, bfr[t].v, (short)0, acc[t], false, false);
    }
  }

  #pragma unroll
  for (int t = 0; t < NT; ++t) {
    const int col = colBase + t * 16 + m;
    #pragma unroll
    for (int i = 0; i < 8; ++i) {
      int rr = rowTile + i + hi * 8;         // C layout: VGPR i -> row i (+8 for hi half)
      if (rr < M) {
        float v = acc[t][i];
        if (rowscale) v *= rowscale[rr];
        if (bias)     v += bias[col];
        if (C16) C16[(size_t)rr * N + col] = f2bf_bits(v);
        else     C  [(size_t)rr * N + col] = v;
      }
    }
  }
}

// ---------- edge scatter-add: one wave per edge, lanes stride features ----------
__global__ void scatter_add_kernel(const long long* __restrict__ src,
                                   const long long* __restrict__ dst,
                                   const float* __restrict__ g,
                                   float* __restrict__ acc, int E, int F) {
  int wid  = (blockIdx.x * blockDim.x + threadIdx.x) >> 5;
  int lane = threadIdx.x & 31;
  if (wid >= E) return;
  int s = (int)src[wid];
  int d = (int)dst[wid];
  const float* gp = g + (size_t)s * F;
  float* ap = acc + (size_t)d * F;
  for (int f = lane; f < F; f += 32) atomicAdd(ap + f, gp[f]);
}

// ---------- post-aggregation: v = act(acc * dinv[row] + bias); write f32 and/or bf16 ----------
__global__ void post_agg_kernel(const float* __restrict__ acc,
                                const float* __restrict__ dinv,
                                const float* __restrict__ bias,
                                float* __restrict__ hf,            // may be null
                                unsigned short* __restrict__ hb,   // may be null
                                int NN, int F, int leaky) {
  int tid = blockIdx.x * blockDim.x + threadIdx.x;
  if (tid >= NN * F) return;
  int n = tid / F, f = tid - n * F;
  float v = acc[tid] * dinv[n] + bias[f];
  if (leaky) v = (v > 0.0f) ? v : 0.01f * v;
  if (hf) hf[tid] = v;
  if (hb) hb[tid] = f2bf_bits(v);
}

// ---------- segment max (encoded atomicMax) ----------
__global__ void segmax_kernel(const float* __restrict__ h,
                              const long long* __restrict__ batch,
                              unsigned* __restrict__ out_enc, int NN, int F) {
  int tid = blockIdx.x * blockDim.x + threadIdx.x;
  if (tid >= NN * F) return;
  int n = tid / F, f = tid - n * F;
  atomicMax(&out_enc[(int)batch[n] * F + f], enc_f(h[tid]));
}

// ---------- fused tiny GEMM (K x NO, NO<=16) + segment max ----------
__global__ void head_segmax_kernel(const float* __restrict__ T,
                                   const float* __restrict__ W,
                                   const float* __restrict__ b,
                                   const long long* __restrict__ batch,
                                   unsigned* __restrict__ out_enc,
                                   int NN, int K, int NO) {
  int n = blockIdx.x * blockDim.x + threadIdx.x;
  if (n >= NN) return;
  float a[16];
  for (int j = 0; j < NO; ++j) a[j] = b[j];
  const float* t = T + (size_t)n * K;
  for (int k = 0; k < K; ++k) {
    float xv = t[k];
    for (int j = 0; j < NO; ++j) a[j] += xv * W[k * NO + j];
  }
  int g = (int)batch[n];
  for (int j = 0; j < NO; ++j) atomicMax(&out_enc[g * NO + j], enc_f(a[j]));
}

// ---------- host launcher ----------
extern "C" void kernel_launch(void* const* d_in, const int* in_sizes, int n_in,
                              void* d_out, int out_size, void* d_ws, size_t ws_size,
                              hipStream_t stream) {
  const float*      x     = (const float*)d_in[0];
  const long long*  edge  = (const long long*)d_in[1];
  const long long*  batch = (const long long*)d_in[2];
  const float* W1 = (const float*)d_in[3];  const float* b1 = (const float*)d_in[4];
  const float* W2 = (const float*)d_in[5];  const float* b2 = (const float*)d_in[6];
  const float* W3 = (const float*)d_in[7];  const float* b3 = (const float*)d_in[8];
  const float* W4 = (const float*)d_in[9];  const float* b4 = (const float*)d_in[10];
  const float* W5 = (const float*)d_in[11]; const float* b5 = (const float*)d_in[12];

  const int NN = in_sizes[2];          // 50000 nodes
  const int E  = in_sizes[1] / 2;      // 400000 edges
  const int F  = in_sizes[0] / NN;     // 128
  const int H1 = in_sizes[4];          // 512
  const int H2 = in_sizes[6];          // 256
  const int H3 = in_sizes[8];          // 128
  const int H4 = in_sizes[10];         // 32
  const int H5 = in_sizes[12];         // 10
  const long long* src = edge;
  const long long* dst = edge + E;

  // workspace layout
  char* base = (char*)d_ws;
  size_t off = 0;
  auto alloc = [&](size_t bytes) -> void* {
    void* p = base + off;
    off = (off + bytes + 255) & ~(size_t)255;
    return p;
  };
  int*            deg  = (int*)alloc((size_t)NN * 4);
  float*          dinv = (float*)alloc((size_t)NN * 4);
  unsigned short* Wt1  = (unsigned short*)alloc((size_t)F  * H1 * 2);
  unsigned short* Wt2  = (unsigned short*)alloc((size_t)H1 * H2 * 2);
  unsigned short* Wt3  = (unsigned short*)alloc((size_t)H2 * H3 * 2);
  unsigned short* Wt4  = (unsigned short*)alloc((size_t)H3 * H4 * 2);
  unsigned short* xbf  = (unsigned short*)alloc((size_t)NN * F  * 2);  // x (bf16), reused as t3bf
  unsigned short* h1bf = (unsigned short*)alloc((size_t)NN * H1 * 2);
  unsigned short* h2bf = (unsigned short*)alloc((size_t)NN * H2 * 2);
  float* R1 = (float*)alloc((size_t)NN * H1 * 4);   // 102.4 MB
  float* R2 = (float*)alloc((size_t)NN * H1 * 4);   // 102.4 MB
  float* R3 = (float*)alloc((size_t)NN * H2 * 4);   //  51.2 MB
  unsigned short* t3bf = xbf;   // x not needed after GEMM1; t3 is [NN,H3], H3<=F

  unsigned* out_u = (unsigned*)d_out;
  const int G = out_size / (H2 + H5);               // 512 graphs

  const int TB = 256;
  auto cdiv = [](long long a, long long b) { return (int)((a + b - 1) / b); };

  // 1) output init (encoded -inf)
  init_enc_kernel<<<cdiv(out_size, TB), TB, 0, stream>>>(out_u, out_size);

  // 2) degrees -> dinv
  deg_init_kernel<<<cdiv(NN, TB), TB, 0, stream>>>(deg, NN);
  deg_count_kernel<<<cdiv(E, TB), TB, 0, stream>>>(dst, deg, E);
  dinv_kernel<<<cdiv(NN, TB), TB, 0, stream>>>(deg, dinv, NN);

  // 3) weights -> bf16 [N][K]; x -> bf16
  transpose_bf16_kernel<<<cdiv((long long)F  * H1, TB), TB, 0, stream>>>(W1, Wt1, F,  H1);
  transpose_bf16_kernel<<<cdiv((long long)H1 * H2, TB), TB, 0, stream>>>(W2, Wt2, H1, H2);
  transpose_bf16_kernel<<<cdiv((long long)H2 * H3, TB), TB, 0, stream>>>(W3, Wt3, H2, H3);
  transpose_bf16_kernel<<<cdiv((long long)H3 * H4, TB), TB, 0, stream>>>(W4, Wt4, H3, H4);
  cvt_bf16_kernel<<<cdiv((long long)NN * F, TB), TB, 0, stream>>>(x, xbf, NN * F);

  // ---- GCN layer 1: g1 = (x @ W1) * dinv[row] ----
  dim3 g1(cdiv(NN, 128), H1 / 64);
  gemm_bf16_kernel<4><<<g1, TB, 0, stream>>>(xbf, Wt1, dinv, nullptr, R1, nullptr, NN, F, H1);
  hipMemcpyAsync(R2, R1, (size_t)NN * H1 * 4, hipMemcpyDeviceToDevice, stream); // self-loop init
  scatter_add_kernel<<<cdiv(E, 8), TB, 0, stream>>>(src, dst, R1, R2, E, H1);
  post_agg_kernel<<<cdiv((long long)NN * H1, TB), TB, 0, stream>>>(
      R2, dinv, b1, nullptr, h1bf, NN, H1, 1);

  // ---- GCN layer 2: g2 = (h1 @ W2) * dinv[row] ----
  dim3 g2(cdiv(NN, 128), H2 / 64);
  gemm_bf16_kernel<4><<<g2, TB, 0, stream>>>(h1bf, Wt2, dinv, nullptr, R3, nullptr, NN, H1, H2);
  hipMemcpyAsync(R2, R3, (size_t)NN * H2 * 4, hipMemcpyDeviceToDevice, stream);
  scatter_add_kernel<<<cdiv(E, 8), TB, 0, stream>>>(src, dst, R3, R2, E, H2);
  post_agg_kernel<<<cdiv((long long)NN * H2, TB), TB, 0, stream>>>(
      R2, dinv, b2, R3, h2bf, NN, H2, 0);

  // ---- embs = segment_max(h2, batch) -> d_out[G*H5 ..] ----
  segmax_kernel<<<cdiv((long long)NN * H2, TB), TB, 0, stream>>>(R3, batch, out_u + G * H5, NN, H2);

  // ---- head: t3 = h2@W3+b3 ; t4 = t3@W4+b4 ; z = t4@W5+b5 ; out = segment_max(z) ----
  dim3 g3(cdiv(NN, 128), H3 / 64);
  gemm_bf16_kernel<4><<<g3, TB, 0, stream>>>(h2bf, Wt3, nullptr, b3, nullptr, t3bf, NN, H2, H3);
  dim3 g4(cdiv(NN, 128), H4 / 32);
  gemm_bf16_kernel<2><<<g4, TB, 0, stream>>>(t3bf, Wt4, nullptr, b4, R1, nullptr, NN, H3, H4);
  head_segmax_kernel<<<cdiv(NN, TB), TB, 0, stream>>>(R1, W5, b5, batch, out_u, NN, H4, H5);

  // 4) decode encoded maxima back to float
  decode_kernel<<<cdiv(out_size, TB), TB, 0, stream>>>(out_u, out_size);
}
